// MultiheadAttn_90726889160825
// MI455X (gfx1250) — matrix operand order repaired and checked by reading, not compiled, not run on hardware
//
#include <hip/hip_runtime.h>
#include <hip/hip_bf16.h>

#define EMBC 1024
#define TSEQ 2048
#define NB   2
#define NH   16
#define HDIM 64

typedef _Float16 v16h __attribute__((ext_vector_type(16)));
typedef _Float16 v8h  __attribute__((ext_vector_type(8)));
typedef _Float16 v4h  __attribute__((ext_vector_type(4)));
typedef float    v8f  __attribute__((ext_vector_type(8)));
typedef int      v4i  __attribute__((vector_size(16)));  // matches builtin sig

#if defined(__gfx1250__) &&                                        \
    __has_builtin(__builtin_amdgcn_global_load_async_to_lds_b128) && \
    __has_builtin(__builtin_amdgcn_s_wait_asynccnt)
#define USE_ASYNC_LDS 1
#else
#define USE_ASYNC_LDS 0
#endif

// Build a v16h A/B fragment from two 16-byte chunks.
static __device__ inline v16h ld16_2(const _Float16* p0, const _Float16* p1) {
  v8h lo = *(const v8h*)p0;
  v8h hi = *(const v8h*)p1;
  return __builtin_shufflevector(lo, hi, 0,1,2,3,4,5,6,7,8,9,10,11,12,13,14,15);
}

static __device__ inline v8f wmma_f16(v16h a, v16h b, v8f c) {
  return __builtin_amdgcn_wmma_f32_16x16x32_f16(false, a, false, b, (short)0, c,
                                                false, false);
}

#if USE_ASYNC_LDS
typedef __attribute__((address_space(1))) v4i* gbl_v4i_p;
typedef __attribute__((address_space(3))) v4i* lds_v4i_p;
// Async-copy 32 bytes global -> LDS (two b128 ops, ASYNCcnt-tracked).
static __device__ inline void async_cp32B(const _Float16* g, _Float16* l) {
  gbl_v4i_p gp = (gbl_v4i_p)g;
  lds_v4i_p lp = (lds_v4i_p)l;
  __builtin_amdgcn_global_load_async_to_lds_b128(gp, lp, 0, 0);
  __builtin_amdgcn_global_load_async_to_lds_b128(gp, lp, 16, 0);
}
#endif

// ---------------------------------------------------------------- conversion
__global__ void cvt_f32_to_f16_kernel(const float* __restrict__ in,
                                      _Float16* __restrict__ out, int n) {
  int i = (blockIdx.x * blockDim.x + threadIdx.x) * 4;
  if (i + 3 < n) {
    float4 f = *(const float4*)(in + i);
    v4h o = {(_Float16)f.x, (_Float16)f.y, (_Float16)f.z, (_Float16)f.w};
    *(v4h*)(out + i) = o;
  }
}

// ---------------------------------------------------------------- WMMA GEMM
// C[M=4096][N=1024] = A[M][K=1024] * Bw[N][K]^T + bias[N]
// MODE 0: f16 out in [B,H,T,hd]    (Q, K)
// MODE 1: f16 out in [B,H,hd,T]    (V transposed; vectorized v8h store)
// MODE 2: f32 out in [M][EMBC]     (final projection -> d_out)
template<int MODE>
__global__ __launch_bounds__(256) void gemm_wmma_kernel(
    const _Float16* __restrict__ A, const _Float16* __restrict__ Bw,
    const float* __restrict__ bias, void* __restrict__ Out) {
  __shared__ alignas(16) _Float16 As[2][128 * 40];  // 40-half pitch: bank-safe
  __shared__ alignas(16) _Float16 Bs[2][128 * 40];  // [N][K] layout

  const int tid  = threadIdx.x;
  const int lane = tid & 31;
  const int wav  = tid >> 5;
  const int wm   = wav & 1;    // 2 wave rows x 64
  const int wn   = wav >> 1;   // 4 wave cols x 32
  const int lg   = lane >> 4;  // half-wave group
  const int ln   = lane & 15;
  const int m0   = blockIdx.y * 128;
  const int n0   = blockIdx.x * 128;

  const int srow  = tid >> 1;        // 128 rows, 2 threads/row
  const int shalf = (tid & 1) * 16;  // 16 halves (32B) per thread

  const _Float16* agBase = A  + (size_t)(m0 + srow) * EMBC + shalf;
  const _Float16* bgBase = Bw + (size_t)(n0 + srow) * EMBC + shalf;
  _Float16* asDst0 = &As[0][srow * 40 + shalf];
  _Float16* bsDst0 = &Bs[0][srow * 40 + shalf];
  _Float16* asDst1 = &As[1][srow * 40 + shalf];
  _Float16* bsDst1 = &Bs[1][srow * 40 + shalf];

  v8f acc[4][2] = {};
  constexpr int NKT = EMBC / 32;  // 32 K-tiles

#if USE_ASYNC_LDS
  // Double-buffered async pipeline: tile i+1 loads overlap tile i compute.
  async_cp32B(agBase, asDst0);
  async_cp32B(bgBase, bsDst0);
  for (int it = 0; it < NKT; ++it) {
    if (it + 1 < NKT) {
      _Float16* ad = (it & 1) ? asDst0 : asDst1;
      _Float16* bd = (it & 1) ? bsDst0 : bsDst1;
      async_cp32B(agBase + (it + 1) * 32, ad);
      async_cp32B(bgBase + (it + 1) * 32, bd);
      __builtin_amdgcn_s_wait_asynccnt(4);  // tile `it` complete (in-order)
    } else {
      __builtin_amdgcn_s_wait_asynccnt(0);
    }
    __syncthreads();
    const int buf = it & 1;
#else
  for (int it = 0; it < NKT; ++it) {
    const int buf = 0;
    *(v8h*)asDst0       = *(const v8h*)(agBase + it * 32);
    *(v8h*)(asDst0 + 8) = *(const v8h*)(agBase + it * 32 + 8);
    *(v8h*)bsDst0       = *(const v8h*)(bgBase + it * 32);
    *(v8h*)(bsDst0 + 8) = *(const v8h*)(bgBase + it * 32 + 8);
    __syncthreads();
#endif

    v16h bf[2];
#pragma unroll
    for (int nt = 0; nt < 2; ++nt) {
      const _Float16* bp = &Bs[buf][(wn * 32 + nt * 16 + ln) * 40 + 16 * lg];
      bf[nt] = ld16_2(bp, bp + 8);
    }
#pragma unroll
    for (int mt = 0; mt < 4; ++mt) {
      const _Float16* ap = &As[buf][(wm * 64 + mt * 16 + ln) * 40];
      v16h af = ld16_2(ap + 16 * lg, ap + 16 * lg + 8);
#pragma unroll
      for (int nt = 0; nt < 2; ++nt)
        acc[mt][nt] = wmma_f16(af, bf[nt], acc[mt][nt]);
    }
    __syncthreads();  // protect next async/sync store into the reused buffer
  }

#pragma unroll
  for (int nt = 0; nt < 2; ++nt) {
    const int col = n0 + wn * 32 + nt * 16 + ln;
    const float bv = bias[col];
#pragma unroll
    for (int mt = 0; mt < 4; ++mt) {
      if (MODE == 1) {
        // [B,H,hd,T]: 8 consecutive t at fixed d -> one 16B store
        const int r0 = m0 + wm * 64 + mt * 16 + 8 * lg;
        const int b = r0 >> 11, t0 = r0 & (TSEQ - 1);
        const int h = col >> 6, d = col & (HDIM - 1);
        v8h pk;
#pragma unroll
        for (int g = 0; g < 8; ++g) pk[g] = (_Float16)(acc[mt][nt][g] + bv);
        *(v8h*)((_Float16*)Out + (((size_t)(b * NH + h)) * HDIM + d) * TSEQ +
                t0) = pk;
      } else {
#pragma unroll
        for (int g = 0; g < 8; ++g) {
          const int row = m0 + wm * 64 + mt * 16 + g + 8 * lg;
          const float v = acc[mt][nt][g] + bv;
          if (MODE == 2) {
            ((float*)Out)[(size_t)row * EMBC + col] = v;
          } else {
            const int b = row >> 11, t = row & (TSEQ - 1);
            const int h = col >> 6,  d = col & (HDIM - 1);
            ((_Float16*)Out)[(((size_t)(b * NH + h)) * TSEQ + t) * HDIM + d] =
                (_Float16)v;
          }
        }
      }
    }
  }
}

// ------------------------------------------------------------ flash attention
// One wave owns 16 query rows; block = 8 waves = 128 rows of one (b,h).
// Streams 32-key chunks with online softmax; O accumulated in f32 WMMA tiles.
__global__ __launch_bounds__(256) void flash_attn_kernel(
    const _Float16* __restrict__ Q, const _Float16* __restrict__ K,
    const _Float16* __restrict__ Vt, _Float16* __restrict__ O) {
  __shared__ alignas(16) _Float16 Pl[8][16 * 40];  // per-wave P staging

  const int lane = threadIdx.x & 31;
  const int wav  = threadIdx.x >> 5;
  const int lg = lane >> 4, ln = lane & 15;
  const int bh = blockIdx.y;
  const int qb = blockIdx.x * 128 + wav * 16;

  const _Float16* qp = Q  + (size_t)bh * TSEQ * HDIM;
  const _Float16* kp = K  + (size_t)bh * TSEQ * HDIM;
  const _Float16* vp = Vt + (size_t)bh * HDIM * TSEQ;

  // Q fragments (16 x 64 = two 16x32 A-fragments), kept in registers.
  v16h qf[2];
#pragma unroll
  for (int ks = 0; ks < 2; ++ks) {
    const _Float16* p = qp + (size_t)(qb + ln) * HDIM + ks * 32;
    qf[ks] = ld16_2(p + 16 * lg, p + 16 * lg + 8);
  }

  v8f o[4] = {};
  float mst[8], lst[8];
#pragma unroll
  for (int g = 0; g < 8; ++g) { mst[g] = -1e30f; lst[g] = 0.f; }

  const int nchunk = (qb + 15) / 32 + 1;  // causal: keys up to qb+15
  for (int kc = 0; kc < nchunk; ++kc) {
    const int key0 = kc * 32;

    if (kc + 1 < nchunk) {  // prefetch next chunk's K/V lines
      __builtin_prefetch(kp + (size_t)(key0 + 32 + ln) * HDIM, 0, 3);
      __builtin_prefetch(vp + (size_t)ln * TSEQ + key0 + 32, 0, 3);
    }

    // S = Q * K^T  (two 16x16 tiles over 32 keys)
    v8f s[2];
#pragma unroll
    for (int nt = 0; nt < 2; ++nt) {
      const _Float16* kb = kp + (size_t)(key0 + nt * 16 + ln) * HDIM;
      v16h b0 = ld16_2(kb + 16 * lg,      kb + 16 * lg + 8);
      v16h b1 = ld16_2(kb + 32 + 16 * lg, kb + 32 + 16 * lg + 8);
      v8f t = {};
      t = wmma_f16(qf[0], b0, t);
      t = wmma_f16(qf[1], b1, t);
      s[nt] = t;
    }

    // scale (hd^-0.5 = 0.125) + causal mask
#pragma unroll
    for (int nt = 0; nt < 2; ++nt)
#pragma unroll
      for (int g = 0; g < 8; ++g) {
        const int qi = qb + g + 8 * lg;
        const int ki = key0 + nt * 16 + ln;
        const float v = s[nt][g] * 0.125f;
        s[nt][g] = (ki > qi) ? -1e30f : v;
      }

    // online softmax: rows live per-VGPR, cols per-lane -> 16-lane reductions
    float sc[8];
#pragma unroll
    for (int g = 0; g < 8; ++g) {
      float rm = fmaxf(s[0][g], s[1][g]);
#pragma unroll
      for (int xm = 1; xm <= 8; xm <<= 1)
        rm = fmaxf(rm, __shfl_xor(rm, xm, 32));
      const float mn = fmaxf(mst[g], rm);
      sc[g] = __expf(mst[g] - mn);
      mst[g] = mn;
      const float p0 = __expf(s[0][g] - mn);
      const float p1 = __expf(s[1][g] - mn);
      float rs = p0 + p1;
#pragma unroll
      for (int xm = 1; xm <= 8; xm <<= 1)
        rs += __shfl_xor(rs, xm, 32);
      lst[g] = lst[g] * sc[g] + rs;
      const int prow = g + 8 * lg;
      Pl[wav][prow * 40 + ln]      = (_Float16)p0;   // C-layout -> LDS
      Pl[wav][prow * 40 + 16 + ln] = (_Float16)p1;
    }
#pragma unroll
    for (int dt = 0; dt < 4; ++dt)
#pragma unroll
      for (int g = 0; g < 8; ++g)
        o[dt][g] *= sc[g];

    // wave-local LDS RAW fence (no block barrier: trip counts diverge)
    asm volatile("s_wait_dscnt 0x0" ::: "memory");

    // P back as a 16x32 A-fragment
    const _Float16* pp = &Pl[wav][ln * 40];
    v16h pf = ld16_2(pp + 16 * lg, pp + 16 * lg + 8);

    // O += P * V   (V transposed [hd][T]: B-fragment = contiguous 32B)
#pragma unroll
    for (int dt = 0; dt < 4; ++dt) {
      const _Float16* vb = vp + (size_t)(dt * 16 + ln) * TSEQ + key0 + 16 * lg;
      v16h vf = ld16_2(vb, vb + 8);
      o[dt] = wmma_f16(pf, vf, o[dt]);
    }
  }

  // normalize and write attention output as f16 [B,T,EMB]
  const int b = bh >> 4, h = bh & (NH - 1);
  float inv[8];
#pragma unroll
  for (int g = 0; g < 8; ++g) inv[g] = 1.0f / lst[g];
#pragma unroll
  for (int dt = 0; dt < 4; ++dt)
#pragma unroll
    for (int g = 0; g < 8; ++g) {
      const int t   = qb + g + 8 * lg;
      const int col = h * HDIM + dt * 16 + ln;
      O[((size_t)b * TSEQ + t) * EMBC + col] = (_Float16)(o[dt][g] * inv[g]);
    }
}

// ---------------------------------------------------------------- launcher
extern "C" void kernel_launch(void* const* d_in, const int* in_sizes, int n_in,
                              void* d_out, int out_size, void* d_ws,
                              size_t ws_size, hipStream_t stream) {
  (void)in_sizes; (void)n_in; (void)out_size; (void)ws_size;
  const float* x  = (const float*)d_in[0];
  const float* Wq = (const float*)d_in[1];
  const float* bq = (const float*)d_in[2];
  const float* Wk = (const float*)d_in[3];
  const float* bk = (const float*)d_in[4];
  const float* Wv = (const float*)d_in[5];
  const float* bv = (const float*)d_in[6];
  const float* Wo = (const float*)d_in[7];
  const float* bo = (const float*)d_in[8];

  const size_t M = (size_t)NB * TSEQ;  // 4096 rows
  _Float16* xh   = (_Float16*)d_ws;    // x in f16; reused as attn output
  _Float16* Wq16 = xh   + M * EMBC;
  _Float16* Wk16 = Wq16 + (size_t)EMBC * EMBC;
  _Float16* Wv16 = Wk16 + (size_t)EMBC * EMBC;
  _Float16* Wo16 = Wv16 + (size_t)EMBC * EMBC;
  _Float16* q16  = Wo16 + (size_t)EMBC * EMBC;  // [B,H,T,hd]
  _Float16* k16  = q16  + M * EMBC;             // [B,H,T,hd]
  _Float16* vt16 = k16  + M * EMBC;             // [B,H,hd,T]

  const int nx = (int)(M * EMBC);
  const int nw = EMBC * EMBC;
  cvt_f32_to_f16_kernel<<<nx / 1024, 256, 0, stream>>>(x,  xh,   nx);
  cvt_f32_to_f16_kernel<<<nw / 1024, 256, 0, stream>>>(Wq, Wq16, nw);
  cvt_f32_to_f16_kernel<<<nw / 1024, 256, 0, stream>>>(Wk, Wk16, nw);
  cvt_f32_to_f16_kernel<<<nw / 1024, 256, 0, stream>>>(Wv, Wv16, nw);
  cvt_f32_to_f16_kernel<<<nw / 1024, 256, 0, stream>>>(Wo, Wo16, nw);

  dim3 ggrid(EMBC / 128, (unsigned)(M / 128));  // (8, 32)
  gemm_wmma_kernel<0><<<ggrid, 256, 0, stream>>>(xh, Wq16, bq, q16);
  gemm_wmma_kernel<0><<<ggrid, 256, 0, stream>>>(xh, Wk16, bk, k16);
  gemm_wmma_kernel<1><<<ggrid, 256, 0, stream>>>(xh, Wv16, bv, vt16);

  dim3 agrid(TSEQ / 128, NB * NH);  // (16, 32)
  flash_attn_kernel<<<agrid, 256, 0, stream>>>(q16, k16, vt16, xh);

  gemm_wmma_kernel<2><<<ggrid, 256, 0, stream>>>(xh, Wo16, bo, d_out);
}
